// RelativeAttention_86517821216489
// MI455X (gfx1250) — matrix operand order repaired and checked by reading, not compiled, not run on hardware
//
#include <hip/hip_runtime.h>
#include <hip/hip_bf16.h>
#include <math.h>
#include <stdint.h>

// ---------------------------------------------------------------------------
// Relative attention, MI455X (gfx1250, wave32, WMMA bf16 w/ fp32 accum).
// Roofline: rel-bias tensor = 512MB read-once -> ~23us HBM floor @23.3TB/s;
// 34 GFLOP in bf16 WMMA ~10-15us -> memory bound. So: bf16 for all matmuls,
// non-temporal streaming of rel, fp32 softmax/accum, async global->LDS DMA
// (ASYNCcnt path) where tiles are already bf16.
// ---------------------------------------------------------------------------

#define N_B   2
#define S_LEN 1024
#define E_DIM 1024
#define H_N   16
#define DK_N  64

typedef __bf16 bf16;
typedef __attribute__((ext_vector_type(16))) __bf16 v16bf;
typedef __attribute__((ext_vector_type(8)))  __bf16 v8bf;
typedef __attribute__((ext_vector_type(8)))  float  v8f;

// ---- gfx1250 async global->LDS DMA (guarded; falls back to VGPR path) ----
#if defined(__has_builtin)
#if __has_builtin(__builtin_amdgcn_global_load_async_to_lds_b128)
#define HAVE_ASYNC_LDS 1
#endif
#endif
#ifndef HAVE_ASYNC_LDS
#define HAVE_ASYNC_LDS 0
#endif

#if HAVE_ASYNC_LDS
typedef int i32x4_g __attribute__((__vector_size__(4 * sizeof(int))));
typedef __attribute__((address_space(1))) i32x4_g* gv4i_ptr;  // global int4*
typedef __attribute__((address_space(3))) i32x4_g* lv4i_ptr;  // LDS int4*

__device__ __forceinline__ void async_copy_b128(const void* gsrc, void* ldst) {
    // generic->AS1: identical bits; generic->AS3: low 32 bits are LDS offset
    __builtin_amdgcn_global_load_async_to_lds_b128(
        (gv4i_ptr)(uintptr_t)gsrc,
        (lv4i_ptr)(unsigned)(uintptr_t)ldst,
        0, 0);
}
__device__ __forceinline__ void wait_async0() {
#if __has_builtin(__builtin_amdgcn_s_wait_asynccnt)
    __builtin_amdgcn_s_wait_asynccnt(0);
#else
    asm volatile("s_wait_asynccnt 0x0" ::: "memory");
#endif
}
#endif

__device__ __forceinline__ v8f zero8() {
    v8f z;
#pragma unroll
    for (int i = 0; i < 8; ++i) z[i] = 0.0f;
    return z;
}

__device__ __forceinline__ v8f wmma_bf16(v16bf a, v16bf b, v8f c) {
    // D = A(16x32 bf16) * B(32x16 bf16) + C(16x16 f32)
    return __builtin_amdgcn_wmma_f32_16x16x32_bf16(false, a, false, b,
                                                   (short)0, c, false, false);
}

__device__ __forceinline__ v16bf join8(v8bf lo, v8bf hi) {
    return __builtin_shufflevector(lo, hi, 0, 1, 2, 3, 4, 5, 6, 7,
                                           8, 9, 10, 11, 12, 13, 14, 15);
}

// 16 contiguous fp32 -> bf16 fragment, streamed with non-temporal loads
// (rel tensor is touched exactly once; keep it out of L2).
__device__ __forceinline__ v16bf cvt16_nt(const float* __restrict__ p) {
    v16bf r;
#pragma unroll
    for (int i = 0; i < 16; ++i) r[i] = (bf16)__builtin_nontemporal_load(p + i);
    return r;
}

// ---------------------------------------------------------------------------
// Kernel 1: projection GEMM  dst = X @ W^T + b  (fp32 in, bf16 out)
// Block: 256 thr (8 waves), 128x128 output tile, K-step 32 via LDS staging.
// mode 0: head-split [n,h,s,dk]   mode 1: transposed head-split [n,h,dk,s]
// ---------------------------------------------------------------------------
__global__ __launch_bounds__(256) void proj_gemm(const float* __restrict__ X,
                                                 const float* __restrict__ W,
                                                 const float* __restrict__ bias,
                                                 bf16* __restrict__ dst,
                                                 int mode) {
    __shared__ alignas(32) bf16 As[128][32];
    __shared__ alignas(32) bf16 BsT[128][32];

    const int bx   = blockIdx.x;           // 16 M-tiles x 8 N-tiles
    const int row0 = (bx >> 3) * 128;
    const int col0 = (bx & 7) * 128;
    const int t    = threadIdx.x;
    const int wid  = t >> 5, lane = t & 31;
    const int waveM = wid >> 2, waveN = wid & 3;
    const int l15 = lane & 15, g = lane >> 4;

    v8f c[4][2];
#pragma unroll
    for (int mi = 0; mi < 4; ++mi)
#pragma unroll
        for (int nj = 0; nj < 2; ++nj) c[mi][nj] = zero8();

    const int r  = t >> 1;          // staging row 0..127
    const int e0 = (t & 1) * 16;    // staging col offset

    for (int kk = 0; kk < E_DIM / 32; ++kk) {
        const float* ap = X + (size_t)(row0 + r) * E_DIM + kk * 32 + e0;
        const float* bp = W + (size_t)(col0 + r) * E_DIM + kk * 32 + e0;
        float4 af[4], bw[4];
#pragma unroll
        for (int i = 0; i < 4; ++i) {
            af[i] = ((const float4*)ap)[i];
            bw[i] = ((const float4*)bp)[i];
        }
        __syncthreads();
#pragma unroll
        for (int i = 0; i < 4; ++i) {
            As[r][e0 + 4 * i + 0] = (bf16)af[i].x;
            As[r][e0 + 4 * i + 1] = (bf16)af[i].y;
            As[r][e0 + 4 * i + 2] = (bf16)af[i].z;
            As[r][e0 + 4 * i + 3] = (bf16)af[i].w;
            BsT[r][e0 + 4 * i + 0] = (bf16)bw[i].x;
            BsT[r][e0 + 4 * i + 1] = (bf16)bw[i].y;
            BsT[r][e0 + 4 * i + 2] = (bf16)bw[i].z;
            BsT[r][e0 + 4 * i + 3] = (bf16)bw[i].w;
        }
        __syncthreads();

        v16bf a[4], b[2];
#pragma unroll
        for (int mi = 0; mi < 4; ++mi) {
            const int ar = waveM * 64 + mi * 16 + l15;
            a[mi] = join8(*(const v8bf*)&As[ar][g * 16],
                          *(const v8bf*)&As[ar][g * 16 + 8]);
        }
#pragma unroll
        for (int nj = 0; nj < 2; ++nj) {
            const int bc = waveN * 32 + nj * 16 + l15;
            b[nj] = *(const v16bf*)&BsT[bc][g * 16];
        }
#pragma unroll
        for (int mi = 0; mi < 4; ++mi)
#pragma unroll
            for (int nj = 0; nj < 2; ++nj)
                c[mi][nj] = wmma_bf16(a[mi], b[nj], c[mi][nj]);
    }

#pragma unroll
    for (int mi = 0; mi < 4; ++mi) {
#pragma unroll
        for (int nj = 0; nj < 2; ++nj) {
            const int col = col0 + waveN * 32 + nj * 16 + l15;
            const float bv = bias[col];
            const int h  = col >> 6;
            const int dk = col & 63;
#pragma unroll
            for (int vv = 0; vv < 8; ++vv) {
                const int row = row0 + waveM * 64 + mi * 16 + vv + 8 * g;
                const float val = c[mi][nj][vv] + bv;
                const int nb = row >> 10, s = row & 1023;
                size_t idx;
                if (mode == 0)
                    idx = (((size_t)nb * H_N + h) * S_LEN + s) * DK_N + dk;
                else
                    idx = (((size_t)nb * H_N + h) * DK_N + dk) * S_LEN + s;
                dst[idx] = (bf16)val;
            }
        }
    }
}

// ---------------------------------------------------------------------------
// Kernel 2: score2[n,h,q,k] = sum_d q[n,h,q,d] * rel[n,q,k,d]
// One block per (n,q): A = Q over heads (16h x 64d) from LDS; B tiles stream
// rel (fp32 -> bf16, non-temporal). 8 waves x 8 k-tiles, 2 WMMA per tile.
// ---------------------------------------------------------------------------
__global__ __launch_bounds__(256) void score2_kernel(const float* __restrict__ rel,
                                                     const bf16* __restrict__ qb,
                                                     bf16* __restrict__ s2) {
    __shared__ alignas(32) bf16 Aq[16][64];
    const int blk = blockIdx.x;        // N*S blocks
    const int n = blk >> 10, qi = blk & 1023;
    const int t = threadIdx.x;

    {   // stage Q-heads tile: thread t -> (h = t>>4, d0 = (t&15)*4)
        const int h = t >> 4, d0 = (t & 15) * 4;
        const bf16* src = qb + (((size_t)n * H_N + h) * S_LEN + qi) * DK_N + d0;
        Aq[h][d0 + 0] = src[0];
        Aq[h][d0 + 1] = src[1];
        Aq[h][d0 + 2] = src[2];
        Aq[h][d0 + 3] = src[3];
    }
    __syncthreads();

    const int wid = t >> 5, lane = t & 31;
    const int l15 = lane & 15, g = lane >> 4;

    v16bf a0, a1;
    {
        const bf16* arow = &Aq[l15][0];   // row m = head
        a0 = join8(*(const v8bf*)&arow[g * 16], *(const v8bf*)&arow[g * 16 + 8]);
        a1 = join8(*(const v8bf*)&arow[32 + g * 16], *(const v8bf*)&arow[40 + g * 16]);
    }

    for (int j = wid; j < 64; j += 8) {
        const int K0 = j * 16;
        const float* rrow =
            rel + (((size_t)n * S_LEN + qi) * S_LEN + K0 + l15) * DK_N + g * 16;
        __builtin_prefetch(rrow + 8 * 16 * DK_N, 0, 0);  // next tile of this wave
        const v16bf b0 = cvt16_nt(rrow);
        const v16bf b1 = cvt16_nt(rrow + 32);
        v8f cacc = zero8();
        cacc = wmma_bf16(a0, b0, cacc);
        cacc = wmma_bf16(a1, b1, cacc);
#pragma unroll
        for (int vv = 0; vv < 8; ++vv) {
            const int hrow = vv + 8 * g;   // D row = head
            s2[(((size_t)n * H_N + hrow) * S_LEN + qi) * S_LEN + K0 + l15] =
                (bf16)cacc[vv];
        }
    }
}

// ---------------------------------------------------------------------------
// Kernel 3: per (n,h,16-q tile): score1 WMMA + add score2, double 1/sqrt(dk)
// scaling (faithful to reference), row softmax in LDS, then P@V WMMA.
// 4 waves/block, one q-tile per wave, 128KB dynamic LDS (bf16 score rows).
// The wave's 16x1024 score2 rows are DMA'd into LDS up front (ASYNCcnt path)
// so the score loop does LDS RMW instead of scattered 2B global loads.
// ---------------------------------------------------------------------------
__global__ __launch_bounds__(128) void attn_kernel(const bf16* __restrict__ qb,
                                                   const bf16* __restrict__ kb,
                                                   const bf16* __restrict__ vbT,
                                                   const bf16* __restrict__ s2,
                                                   bf16* __restrict__ ctx) {
    extern __shared__ char smem[];
    const int t = threadIdx.x, wid = t >> 5, lane = t & 31;
    const int l15 = lane & 15, g = lane >> 4;

    bf16*  sc = (bf16*)smem + (size_t)wid * 16 * 1024;         // 32KB / wave
    float* rs = (float*)(smem + 4 * 16 * 1024 * 2) + wid * 16; // 1/rowsum

    const int tile = blockIdx.x * 4 + wid;     // 2048 q-tiles total
    const int n = tile >> 10;
    const int rem = tile & 1023;
    const int h = rem >> 6;
    const int Q0 = (rem & 63) << 4;

    const float inv = 0.125f;   // 1/sqrt(64)

    const size_t hb = ((size_t)n * H_N + h) * S_LEN;

#if HAVE_ASYNC_LDS
    // Bulk-copy this wave's score2 tile rows (16 x 2KB) into LDS via DMA.
    for (int i = lane; i < 16 * 128; i += 32) {      // 2048 x 16B chunks
        const int m  = i >> 7;                       // row 0..15
        const int co = (i & 127) * 8;                // element offset
        async_copy_b128(s2 + (hb + Q0 + m) * S_LEN + co, &sc[m * 1024 + co]);
    }
#endif

    // A fragments: Q tile rows (contraction d contiguous)
    const bf16* qrow = qb + (((size_t)n * H_N + h) * S_LEN + Q0 + l15) * DK_N;
    const v16bf a0 = join8(*(const v8bf*)&qrow[g * 16], *(const v8bf*)&qrow[g * 16 + 8]);
    const v16bf a1 = join8(*(const v8bf*)&qrow[32 + g * 16], *(const v8bf*)&qrow[40 + g * 16]);

#if HAVE_ASYNC_LDS
    wait_async0();   // wave-private LDS region; DS stays in order afterwards
#endif

    // ---- scores = (QK^T/sqrt(dk) + score2)/sqrt(dk) ----
    for (int j = 0; j < 64; ++j) {
        const int K0 = j * 16;
        const bf16* krow = kb + (hb + K0 + l15) * DK_N;
        const v16bf b0 = *(const v16bf*)&krow[g * 16];
        const v16bf b1 = *(const v16bf*)&krow[32 + g * 16];
        v8f cacc = zero8();
        cacc = wmma_bf16(a0, b0, cacc);
        cacc = wmma_bf16(a1, b1, cacc);
#pragma unroll
        for (int vv = 0; vv < 8; ++vv) {
            const int m = vv + 8 * g;
#if HAVE_ASYNC_LDS
            const float s2v = (float)sc[m * 1024 + K0 + l15];
#else
            const float s2v = (float)s2[(hb + Q0 + m) * S_LEN + K0 + l15];
#endif
            const float sv = (cacc[vv] * inv + s2v) * inv;
            sc[m * 1024 + K0 + l15] = (bf16)sv;
        }
    }
    __syncthreads();

    // ---- row softmax (store unnormalized exp, keep 1/sum in rs) ----
    {
        const int rrow = lane >> 1, hf = lane & 1;
        const int base = rrow * 1024 + hf * 512;
        float mx = -1e30f;
#pragma unroll 8
        for (int i = 0; i < 512; ++i) mx = fmaxf(mx, (float)sc[base + i]);
        mx = fmaxf(mx, __shfl_xor(mx, 1, 32));
        float sum = 0.0f;
#pragma unroll 8
        for (int i = 0; i < 512; ++i) {
            const float e = __expf((float)sc[base + i] - mx);
            sum += e;
            sc[base + i] = (bf16)e;
        }
        sum += __shfl_xor(sum, 1, 32);
        if (hf == 0) rs[rrow] = 1.0f / sum;
    }
    __syncthreads();

    // ---- O = P @ V (V stored transposed: contraction s contiguous) ----
    v8f o[4];
#pragma unroll
    for (int d = 0; d < 4; ++d) o[d] = zero8();

    const size_t vb0 = ((size_t)n * H_N + h) * DK_N;
    for (int jj = 0; jj < 32; ++jj) {
        const bf16* prow = &sc[l15 * 1024 + jj * 32];
        const v16bf ap = join8(*(const v8bf*)&prow[g * 16],
                               *(const v8bf*)&prow[g * 16 + 8]);
#pragma unroll
        for (int d = 0; d < 4; ++d) {
            const int dcol = d * 16 + l15;
            const bf16* vrow = vbT + (vb0 + dcol) * S_LEN + jj * 32 + g * 16;
            o[d] = wmma_bf16(ap, *(const v16bf*)vrow, o[d]);
        }
    }

#pragma unroll
    for (int d = 0; d < 4; ++d) {
#pragma unroll
        for (int vv = 0; vv < 8; ++vv) {
            const int m = vv + 8 * g;
            const float val = o[d][vv] * rs[m];
            ctx[((size_t)n * S_LEN + Q0 + m) * E_DIM + h * 64 + d * 16 + l15] =
                (bf16)val;
        }
    }
}

// ---------------------------------------------------------------------------
// Kernel 4: output projection  out = ctx @ Wo^T + bo  (bf16 A, fp32 out)
// A tile (already bf16) staged into LDS via async DMA when available.
// ---------------------------------------------------------------------------
__global__ __launch_bounds__(256) void out_gemm(const bf16* __restrict__ ctx,
                                                const float* __restrict__ Wo,
                                                const float* __restrict__ bo,
                                                float* __restrict__ out) {
    __shared__ alignas(32) bf16 As[128][32];
    __shared__ alignas(32) bf16 BsT[128][32];

    const int bx   = blockIdx.x;
    const int row0 = (bx >> 3) * 128;
    const int col0 = (bx & 7) * 128;
    const int t    = threadIdx.x;
    const int wid  = t >> 5, lane = t & 31;
    const int waveM = wid >> 2, waveN = wid & 3;
    const int l15 = lane & 15, g = lane >> 4;

    v8f c[4][2];
#pragma unroll
    for (int mi = 0; mi < 4; ++mi)
#pragma unroll
        for (int nj = 0; nj < 2; ++nj) c[mi][nj] = zero8();

    const int r  = t >> 1;
    const int e0 = (t & 1) * 16;

    for (int kk = 0; kk < E_DIM / 32; ++kk) {
        const bf16* ap = ctx + (size_t)(row0 + r) * E_DIM + kk * 32 + e0;
        const float* bp = Wo + (size_t)(col0 + r) * E_DIM + kk * 32 + e0;
        float4 bw[4];
#pragma unroll
        for (int i = 0; i < 4; ++i) bw[i] = ((const float4*)bp)[i];
#if !HAVE_ASYNC_LDS
        const v8bf alo = ((const v8bf*)ap)[0];
        const v8bf ahi = ((const v8bf*)ap)[1];
#endif
        __syncthreads();   // previous tile's fragment reads retired
#if HAVE_ASYNC_LDS
        async_copy_b128(ap,     &As[r][e0]);       // direct DMA, no VGPR hop
        async_copy_b128(ap + 8, &As[r][e0 + 8]);
#else
        *(v8bf*)&As[r][e0]     = alo;
        *(v8bf*)&As[r][e0 + 8] = ahi;
#endif
#pragma unroll
        for (int i = 0; i < 4; ++i) {
            BsT[r][e0 + 4 * i + 0] = (bf16)bw[i].x;
            BsT[r][e0 + 4 * i + 1] = (bf16)bw[i].y;
            BsT[r][e0 + 4 * i + 2] = (bf16)bw[i].z;
            BsT[r][e0 + 4 * i + 3] = (bf16)bw[i].w;
        }
#if HAVE_ASYNC_LDS
        wait_async0();
#endif
        __syncthreads();

        v16bf a[4], b[2];
#pragma unroll
        for (int mi = 0; mi < 4; ++mi) {
            const int ar = waveM * 64 + mi * 16 + l15;
            a[mi] = join8(*(const v8bf*)&As[ar][g * 16],
                          *(const v8bf*)&As[ar][g * 16 + 8]);
        }
#pragma unroll
        for (int nj = 0; nj < 2; ++nj) {
            const int bc = waveN * 32 + nj * 16 + l15;
            b[nj] = *(const v16bf*)&BsT[bc][g * 16];
        }
#pragma unroll
        for (int mi = 0; mi < 4; ++mi)
#pragma unroll
            for (int nj = 0; nj < 2; ++nj)
                c[mi][nj] = wmma_bf16(a[mi], b[nj], c[mi][nj]);
    }

#pragma unroll
    for (int mi = 0; mi < 4; ++mi) {
#pragma unroll
        for (int nj = 0; nj < 2; ++nj) {
            const int col = col0 + waveN * 32 + nj * 16 + l15;
            const float bv = bo[col];
#pragma unroll
            for (int vv = 0; vv < 8; ++vv) {
                const int row = row0 + waveM * 64 + mi * 16 + vv + 8 * g;
                out[(size_t)row * E_DIM + col] = c[mi][nj][vv] + bv;
            }
        }
    }
}

// ---------------------------------------------------------------------------
extern "C" void kernel_launch(void* const* d_in, const int* in_sizes, int n_in,
                              void* d_out, int out_size, void* d_ws, size_t ws_size,
                              hipStream_t stream) {
    const float* query = (const float*)d_in[0];
    const float* key_  = (const float*)d_in[1];
    const float* value = (const float*)d_in[2];
    const float* rel   = (const float*)d_in[3];
    const float* Wq = (const float*)d_in[4];
    const float* bq = (const float*)d_in[5];
    const float* Wk = (const float*)d_in[6];
    const float* bk = (const float*)d_in[7];
    const float* Wv = (const float*)d_in[8];
    const float* bv = (const float*)d_in[9];
    const float* Wo = (const float*)d_in[10];
    const float* bo = (const float*)d_in[11];
    float* out = (float*)d_out;

    // Workspace carve-up (all offsets 256B aligned):
    //   qb,kb,vbT,ctx: 4MB bf16 each; s2: 64MB bf16  -> 80MB total
    char* ws = (char*)d_ws;
    const size_t MB4 = (size_t)4 << 20;
    bf16* qb  = (bf16*)(ws);
    bf16* kb  = (bf16*)(ws + MB4);
    bf16* vbT = (bf16*)(ws + 2 * MB4);
    bf16* ctx = (bf16*)(ws + 3 * MB4);
    bf16* s2  = (bf16*)(ws + 4 * MB4);

    proj_gemm<<<128, 256, 0, stream>>>(query, Wq, bq, qb, 0);
    proj_gemm<<<128, 256, 0, stream>>>(key_,  Wk, bk, kb, 0);
    proj_gemm<<<128, 256, 0, stream>>>(value, Wv, bv, vbT, 1);

    score2_kernel<<<N_B * S_LEN, 256, 0, stream>>>(rel, qb, s2);

    const size_t attn_lds = 4 * 16 * 1024 * 2 /*scores*/ + 4 * 16 * 4 /*rowsums*/;
    attn_kernel<<<512, 128, attn_lds, stream>>>(qb, kb, vbT, s2, ctx);

    out_gemm<<<128, 256, 0, stream>>>(ctx, Wo, bo, out);
}
